// IncrSeqAttnMatch_86844238725181
// MI455X (gfx1250) — matrix operand order repaired and checked by reading, not compiled, not run on hardware
//
#include <hip/hip_runtime.h>
#include <hip/hip_bf16.h>
#include <math.h>

// ---------------------------------------------------------------------------
// Problem constants (reference: T=128, LQ=LA=64, H=1024, HID=250)
// ---------------------------------------------------------------------------
#define TT   128
#define LQ   64
#define LT   128                 // tokens appended per turn (LQ + LA)
#define MM   (TT * LT)           // 16384 memory slots
#define HH   1024                // embedding dim
#define HIDP 256                 // HID=250 padded to 256 (zero pad -> contributes 0)

typedef __bf16 bf16_t;
typedef __attribute__((ext_vector_type(8)))  __bf16 v8bf;
typedef __attribute__((ext_vector_type(16))) __bf16 v16bf;
typedef __attribute__((ext_vector_type(8)))  float  v8f;

union V16U { v16bf v; v8bf h[2]; };

__device__ __forceinline__ v8f wmma_bf16(v16bf a, v16bf b, v8f c) {
  // D = A(16x32 bf16) x B(32x16 bf16) + C(16x16 f32)
  return __builtin_amdgcn_wmma_f32_16x16x32_bf16(
      /*neg_a=*/false, a, /*neg_b=*/false, b,
      /*c_mod=*/(short)0, c, /*reuse_a=*/false, /*reuse_b=*/false);
}

// A-operand 16x32 bf16 tile from a row-major [rows, stride] bf16 matrix.
// lane L holds matrix row (row0 + (L&15)); per-ISA K interleave:
//   lanes 0-15 : K = kk+[0..7] and kk+16+[0..7]
//   lanes16-31 : K = kk+[8..15] and kk+24+[0..7]
__device__ __forceinline__ v16bf load_a_tile(const bf16_t* base, int stride,
                                             int row0, int kk, int lane) {
  int r  = row0 + (lane & 15);
  int hl = lane >> 4;
  const bf16_t* p = base + (size_t)r * stride + kk;
  V16U t;
  t.h[0] = *(const v8bf*)(p + hl * 8);
  t.h[1] = *(const v8bf*)(p + 16 + hl * 8);
  return t.v;
}

// B-operand 32x16 bf16 tile: B[k, n] = src[col0 + n][kk + k] (src row-major).
// lane L holds B column (col0 + (L&15)); lanes 0-15 carry K 0..7,16..23,
// lanes 16-31 carry K 8..15,24..31 (two 16B chunks per lane).
__device__ __forceinline__ v16bf load_b_tile(const bf16_t* base, size_t stride,
                                             int col0, int kk, int lane) {
  int c  = col0 + (lane & 15);
  int hl = lane >> 4;
  const bf16_t* p = base + (size_t)c * stride + kk + hl * 8;
  V16U t;
  t.h[0] = *(const v8bf*)(p);
  t.h[1] = *(const v8bf*)(p + 16);
  return t.v;
}

__device__ __forceinline__ v8f vzero8() {
  return (v8f){0.f, 0.f, 0.f, 0.f, 0.f, 0.f, 0.f, 0.f};
}

// ---------------------------------------------------------------------------
// Kernel 1: bf16 staging.  X_bf[m, c] and VT[c, m] (transposed values) from
// the f32 embeddings; memory layout m = t*128 + slot (slot<64: q, else a).
// Also emits turn-0 output rows (xq_plus[0] == xq_emb[0]).
// ---------------------------------------------------------------------------
__global__ void k_prep(const float* __restrict__ xq, const float* __restrict__ xa,
                       float* __restrict__ out, bf16_t* __restrict__ Xbf,
                       bf16_t* __restrict__ VT) {
  size_t idx = (size_t)blockIdx.x * blockDim.x + threadIdx.x;
  if (idx >= (size_t)MM * HH) return;
  int m = (int)(idx >> 10);
  int c = (int)(idx & (HH - 1));
  int t = m >> 7, slot = m & 127;
  float v = (slot < LQ) ? xq[((size_t)t * LQ + slot) * HH + c]
                        : xa[((size_t)t * LQ + (slot - LQ)) * HH + c];
  bf16_t bv = (bf16_t)v;
  Xbf[idx] = bv;
  VT[(size_t)c * MM + m] = bv;
  if (t == 0 && slot < LQ) out[(size_t)slot * HH + c] = v;  // turn 0 passthrough
}

// ---------------------------------------------------------------------------
// Kernel 2: W (HID x H f32) -> bf16 [256, 1024], rows 250..255 zero.
// ---------------------------------------------------------------------------
__global__ void k_wpack(const float* __restrict__ W, bf16_t* __restrict__ Wbf) {
  int idx = blockIdx.x * blockDim.x + threadIdx.x;   // 256*1024 threads
  int n = idx >> 10, k = idx & (HH - 1);
  float v = (n < 250) ? W[(size_t)n * HH + k] : 0.0f;
  Wbf[idx] = (bf16_t)v;
}

// ---------------------------------------------------------------------------
// Kernel 3: projection GEMM.  memproj[16384, 256] = relu(X_bf @ Wbf^T + b).
// 1024 waves, each owns one 16-row M tile.  N is processed in two passes of
// 8 accumulators (64 VGPRs) so the K loop never spills: per k-step one A
// load is streamed against 8 B tiles.
// ---------------------------------------------------------------------------
__global__ void k_proj(const bf16_t* __restrict__ Xbf, const bf16_t* __restrict__ Wbf,
                       const float* __restrict__ bias, bf16_t* __restrict__ memproj) {
  int wave = blockIdx.x * (blockDim.x >> 5) + (threadIdx.x >> 5);
  int lane = threadIdx.x & 31;
  int hl = lane >> 4, l16 = lane & 15;
  int m0 = wave * 16;

#pragma unroll 1
  for (int nb = 0; nb < 2; ++nb) {
    v8f acc[8];
#pragma unroll
    for (int j = 0; j < 8; ++j) acc[j] = vzero8();

#pragma unroll 1
    for (int kk = 0; kk < HH; kk += 32) {
      v16bf a = load_a_tile(Xbf, HH, m0, kk, lane);
#pragma unroll
      for (int j = 0; j < 8; ++j) {
        v16bf b = load_b_tile(Wbf, HH, (nb * 8 + j) * 16, kk, lane);
        acc[j] = wmma_bf16(a, b, acc[j]);
      }
    }

#pragma unroll
    for (int j = 0; j < 8; ++j) {
      int n = (nb * 8 + j) * 16 + l16;
      float bv = (n < 250) ? bias[n] : 0.0f;
#pragma unroll
      for (int r = 0; r < 8; ++r) {
        float v = acc[j][r] + bv;
        v = v > 0.0f ? v : 0.0f;                     // relu
        int row = m0 + r + hl * 8;
        memproj[(size_t)row * HIDP + n] = (bf16_t)v;
      }
    }
  }
}

// ---------------------------------------------------------------------------
// Kernel 4: static softmax statistics.  One block per turn t (t = bid+1).
// Q k-step tiles preloaded to registers (invariant over the whole kernel).
// Streams the [64, t*128] score matrix in 128-column chunks through LDS,
// maintaining per-row running (max, sum-exp) with padding-mask handling.
// Score N-tiles computed in two groups of 4 to bound register pressure.
// ---------------------------------------------------------------------------
__global__ void k_stats(const bf16_t* __restrict__ memproj,
                        const unsigned char* __restrict__ qmask,
                        const unsigned char* __restrict__ amask,
                        float* __restrict__ rowm, float* __restrict__ rowl) {
  __shared__ float S[64 * 128];                      // 32 KB
  int t = blockIdx.x + 1;
  int tid = threadIdx.x, wave = tid >> 5, lane = tid & 31;
  int hl = lane >> 4, l16 = lane & 15;
  const bf16_t* Q = memproj + (size_t)(t * LT) * HIDP;

  // Preload this wave's Q row-block (16 rows x K=256) into 8 A tiles.
  v16bf qt[8];
#pragma unroll
  for (int ks = 0; ks < 8; ++ks)
    qt[ks] = load_a_tile(Q, HIDP, wave * 16, ks * 32, lane);

  float rm = -3.0e38f, rl = 0.0f;
  for (int s = 0; s < t; ++s) {
    // 4 waves x 8 N-tiles = full [64,128] score chunk vs memory turn s.
#pragma unroll 1
    for (int g = 0; g < 2; ++g) {
      v8f sc[4];
#pragma unroll
      for (int j = 0; j < 4; ++j) sc[j] = vzero8();
#pragma unroll
      for (int ks = 0; ks < 8; ++ks) {
#pragma unroll
        for (int j = 0; j < 4; ++j) {
          v16bf b = load_b_tile(memproj, HIDP,
                                s * LT + (g * 4 + j) * 16, ks * 32, lane);
          sc[j] = wmma_bf16(qt[ks], b, sc[j]);
        }
      }
#pragma unroll
      for (int j = 0; j < 4; ++j) {
        int col = (g * 4 + j) * 16 + l16;
#pragma unroll
        for (int r = 0; r < 8; ++r)
          S[(wave * 16 + r + hl * 8) * 128 + col] = sc[j][r];
      }
    }
    __syncthreads();
    if (tid < 64) {
      float chmax = -3.0e38f;
      for (int c = 0; c < 128; ++c) {
        bool msk = (c < LQ) ? (qmask[s * LQ + c] != 0)
                            : (amask[s * LQ + (c - LQ)] != 0);
        float v = msk ? -3.0e38f : S[tid * 128 + c];
        chmax = fmaxf(chmax, v);
      }
      if (chmax > -1.0e37f) {
        float nm = fmaxf(rm, chmax);
        float acc = rl * __expf(rm - nm);
        for (int c = 0; c < 128; ++c) {
          bool msk = (c < LQ) ? (qmask[s * LQ + c] != 0)
                              : (amask[s * LQ + (c - LQ)] != 0);
          if (!msk) acc += __expf(S[tid * 128 + c] - nm);
        }
        rm = nm; rl = acc;
      }
    }
    __syncthreads();
  }
  if (tid < 64) { rowm[t * 64 + tid] = rm; rowl[t * 64 + tid] = rl; }
}

// ---------------------------------------------------------------------------
// Kernel 5: sequential scan step (launched once per t, stream-ordered).
// Grid: 8 blocks x 128 threads (4 waves). Block bc owns hist columns
// [bc*128, bc*128+128). Each block recomputes the probability tile
// P = exp(S - m)/l (cheap: K=256) in LDS, then WMMA-accumulates P @ V
// over its column slice (V read from transposed bf16 values VT).
// Epilogue: out = 0.5*xq_emb[t] + 0.5*hist; scatter bf16 into VT for
// consumption by later steps.
// ---------------------------------------------------------------------------
__global__ void k_hist(int t, const bf16_t* __restrict__ memproj,
                       const bf16_t* __restrict__ VT,
                       const float* __restrict__ rowm, const float* __restrict__ rowl,
                       const unsigned char* __restrict__ qmask,
                       const unsigned char* __restrict__ amask,
                       const float* __restrict__ xq,
                       float* __restrict__ out, bf16_t* __restrict__ VTmut) {
  __shared__ bf16_t P[64 * 128];                     // 16 KB
  int bc = blockIdx.x;                               // 0..7 -> hist col slice
  int tid = threadIdx.x, wave = tid >> 5, lane = tid & 31;
  int hl = lane >> 4, l16 = lane & 15;
  const bf16_t* Q = memproj + (size_t)(t * LT) * HIDP;

  // Wave owns score rows [wave*16, wave*16+16): preload Q tiles (64 VGPRs).
  v16bf qt[8];
#pragma unroll
  for (int ks = 0; ks < 8; ++ks)
    qt[ks] = load_a_tile(Q, HIDP, wave * 16, ks * 32, lane);

  // Per-row softmax stats for this wave's 16 score rows (8 rows per lane).
  int srow0 = wave * 16 + hl * 8;                    // rows srow0 .. srow0+7

  // Hist accumulators: wave owns 2 N-tiles (32 cols) x 4 M-tiles.
  v8f acc[4][2];
#pragma unroll
  for (int i = 0; i < 4; ++i) {
    acc[i][0] = vzero8();
    acc[i][1] = vzero8();
  }
  int vcol0 = bc * 128 + wave * 32;
  int vcol1 = vcol0 + 16;

  for (int s = 0; s < t; ++s) {
    // ---- Phase 1: P rows [wave*16, +16), two groups of 4 N-tiles -------
#pragma unroll 1
    for (int g = 0; g < 2; ++g) {
      v8f sc[4];
#pragma unroll
      for (int j = 0; j < 4; ++j) sc[j] = vzero8();
#pragma unroll
      for (int ks = 0; ks < 8; ++ks) {
#pragma unroll
        for (int j = 0; j < 4; ++j) {
          v16bf b = load_b_tile(memproj, HIDP,
                                s * LT + (g * 4 + j) * 16, ks * 32, lane);
          sc[j] = wmma_bf16(qt[ks], b, sc[j]);
        }
      }
#pragma unroll
      for (int j = 0; j < 4; ++j) {
        int col = (g * 4 + j) * 16 + l16;
        bool msk = (col < LQ) ? (qmask[s * LQ + col] != 0)
                              : (amask[s * LQ + (col - LQ)] != 0);
#pragma unroll
        for (int r = 0; r < 8; ++r) {
          int row = srow0 + r;
          float p = 0.0f;
          if (!msk) {
            float l = fmaxf(rowl[t * 64 + row], 1e-30f);
            p = __expf(sc[j][r] - rowm[t * 64 + row]) / l;
          }
          P[row * 128 + col] = (bf16_t)p;
        }
      }
    }
    __syncthreads();
    // ---- Phase 2: hist += P @ V  (K = 128 memory rows of turn s) -------
#pragma unroll
    for (int kk = 0; kk < 128; kk += 32) {
      v16bf b0 = load_b_tile(VT, (size_t)MM, vcol0, s * LT + kk, lane);
      v16bf b1 = load_b_tile(VT, (size_t)MM, vcol1, s * LT + kk, lane);
#pragma unroll
      for (int mt = 0; mt < 4; ++mt) {
        v16bf a = load_a_tile(P, 128, mt * 16, kk, lane);   // from LDS
        acc[mt][0] = wmma_bf16(a, b0, acc[mt][0]);
        acc[mt][1] = wmma_bf16(a, b1, acc[mt][1]);
      }
    }
    __syncthreads();
  }

  // ---- Epilogue: merge + write f32 output + bf16 transposed values ----
#pragma unroll
  for (int mt = 0; mt < 4; ++mt)
#pragma unroll
    for (int nt = 0; nt < 2; ++nt) {
      int col = vcol0 + nt * 16 + l16;                      // hidden dim col
#pragma unroll
      for (int r = 0; r < 8; ++r) {
        int row = mt * 16 + r + hl * 8;                     // q row 0..63
        float v = 0.5f * xq[((size_t)t * LQ + row) * HH + col]
                + 0.5f * acc[mt][nt][r];
        out[((size_t)t * LQ + row) * HH + col] = v;
        VTmut[(size_t)col * MM + (size_t)(t * LT + row)] = (bf16_t)v;
      }
    }
}

// ---------------------------------------------------------------------------
// Host side
// ---------------------------------------------------------------------------
extern "C" void kernel_launch(void* const* d_in, const int* in_sizes, int n_in,
                              void* d_out, int out_size, void* d_ws, size_t ws_size,
                              hipStream_t stream) {
  const float* xq = (const float*)d_in[0];
  const float* xa = (const float*)d_in[1];
  const unsigned char* qmask = (const unsigned char*)d_in[2];
  const unsigned char* amask = (const unsigned char*)d_in[3];
  const float* W = (const float*)d_in[4];
  const float* b = (const float*)d_in[5];
  float* out = (float*)d_out;

  // workspace layout (~76 MB)
  char* w = (char*)d_ws;
  bf16_t* Xbf = (bf16_t*)w;      w += (size_t)MM * HH * 2;     // 33.5 MB
  bf16_t* VT  = (bf16_t*)w;      w += (size_t)HH * MM * 2;     // 33.5 MB
  bf16_t* Wbf = (bf16_t*)w;      w += (size_t)HIDP * HH * 2;   // 0.5 MB
  bf16_t* MP  = (bf16_t*)w;      w += (size_t)MM * HIDP * 2;   // 8.4 MB
  float* rowm = (float*)w;       w += (size_t)TT * 64 * 4;
  float* rowl = (float*)w;       w += (size_t)TT * 64 * 4;

  // Phase 0: bf16 staging + turn-0 output
  k_prep<<<(MM * HH) / 256, 256, 0, stream>>>(xq, xa, out, Xbf, VT);
  k_wpack<<<(HIDP * HH) / 256, 256, 0, stream>>>(W, Wbf);
  // Phase 1: projection GEMM (1024 waves)
  k_proj<<<128, 256, 0, stream>>>(Xbf, Wbf, b, MP);
  // Phase 2: static softmax stats, parallel over turns
  k_stats<<<TT - 1, 128, 0, stream>>>(MP, qmask, amask, rowm, rowl);
  // Phase 3: sequential scan (stream order enforces t-dependency)
  for (int t = 1; t < TT; ++t)
    k_hist<<<8, 128, 0, stream>>>(t, MP, VT, rowm, rowl, qmask, amask,
                                  xq, out, VT);
}
